// HSAttention_15616501088810
// MI455X (gfx1250) — matrix operand order repaired and checked by reading, not compiled, not run on hardware
//
#include <hip/hip_runtime.h>

// ---------------------------------------------------------------------------
// HSAttention (idx=0 branch) for MI455X / gfx1250, wave32 + WMMA bf16.
// Grid: (64 window-batches, 8 heads, 8 row-tiles of 64). Block: 256 = 8 waves.
// ---------------------------------------------------------------------------

typedef __bf16 bf16_t;
typedef __attribute__((ext_vector_type(16))) __bf16 v16bf;
typedef __attribute__((ext_vector_type(8)))  __bf16 v8bf;
typedef __attribute__((ext_vector_type(4)))  __bf16 v4bf;
typedef __attribute__((ext_vector_type(8)))  float  v8f;
typedef __attribute__((ext_vector_type(4)))  float  v4f;

#define B_      8
#define DIM_    512
#define HEADS_  8
#define HD_     64          // head dim
#define HIMG    64
#define WIMG    64
#define SPLIT_  8
#define S_      512         // tokens per window
#define MROWS   64          // score rows per block
#define QS_STR  72          // Qs row stride (bf16), padded
#define KS_STR  72          // Ks row stride (bf16), padded
#define VS_STR  520         // Vst row stride (bf16), padded
#define SC_STR  520         // score row stride (f32), padded
#define AT_STR  520         // attn row stride (bf16), padded
#define PI_HALF 1.57079632679489662f

// LDS budget (bytes):
//   Qs  64*72*2   =   9216
//   Ks  512*72*2  =  73728   (reused as attn: 64*520*2 = 66560)
//   Vst 64*520*2  =  66560
//   score 64*520*4= 133120
//   phi 512*4     =   2048
//   l1p 256*4     =   1024
//   l1f 64*4      =    256
//   cw  576*4     =   2304
//   cb  64*4      =    256
#define SMEM_BYTES 288512

union Frag { v16bf v; v8bf h[2]; };

__device__ __forceinline__ v8f wmma_bf16(const Frag& a, const Frag& b, v8f c) {
  // D = A(16x32 bf16) * B(32x16 bf16) + C(16x16 f32)
  return __builtin_amdgcn_wmma_f32_16x16x32_bf16(false, a.v, false, b.v,
                                                 (short)0, c, false, false);
}

extern "C" __global__ __launch_bounds__(256, 1)
void hsattn_cdna5(const float* __restrict__ temp,
                  const float* __restrict__ polar,
                  const float* __restrict__ convw,
                  const float* __restrict__ convb,
                  float* __restrict__ out)
{
  extern __shared__ char smem[];
  bf16_t* Qs   = (bf16_t*)smem;                   // [64][72]
  bf16_t* Ks   = Qs  + MROWS * QS_STR;            // [512][72] token-major
  bf16_t* Vst  = Ks  + S_   * KS_STR;             // [64][520] dch-major (transposed)
  float*  score= (float*)(Vst + HD_ * VS_STR);    // [64][520]
  float*  phi  = score + MROWS * SC_STR;          // [512]
  float*  l1p  = phi   + S_;                      // [4][64]
  float*  l1f  = l1p   + 4 * MROWS;               // [64]
  float*  cw   = l1f   + MROWS;                   // [64][9]
  float*  cb   = cw    + HD_ * 9;                 // [64]
  bf16_t* attn = Ks;                              // reuse after barrier: [64][520]

  const int tid  = threadIdx.x;
  const int lane = tid & 31;
  const int wv   = tid >> 5;

  const int b    = blockIdx.x >> 3;     // batch
  const int nw   = blockIdx.x & 7;      // vertical stripe index
  const int head = blockIdx.y;
  const int m0   = blockIdx.z * MROWS;  // first score row (token) of this block

  const size_t img = (size_t)HIMG * WIMG;                     // 4096
  const float* qg = temp;
  const float* kg = temp +     (size_t)B_ * DIM_ * img;
  const float* vg = temp + 2 * (size_t)B_ * DIM_ * img;
  const size_t cbase = ((size_t)b * DIM_ + (size_t)head * HD_) * img;

  // ---------------- Phase 1: stage inputs to LDS (single HBM pass) ---------
  for (int e = tid; e < MROWS * HD_; e += 256) {              // Q stripe, *scale
    int dch = e >> 6, sl = e & 63;
    int s = m0 + sl, h = s >> 3, w = nw * SPLIT_ + (s & 7);
    Qs[sl * QS_STR + dch] =
        (bf16_t)(qg[cbase + (size_t)dch * img + h * WIMG + w] * 0.125f);
  }
  for (int e = tid; e < S_ * HD_; e += 256) {                 // K token-major
    int dch = e >> 9, tok = e & 511;
    int h = tok >> 3, w = nw * SPLIT_ + (tok & 7);
    Ks[tok * KS_STR + dch] =
        (bf16_t)kg[cbase + (size_t)dch * img + h * WIMG + w];
  }
  for (int e = tid; e < S_ * HD_; e += 256) {                 // V transposed
    int dch = e >> 9, tok = e & 511;
    int h = tok >> 3, w = nw * SPLIT_ + (tok & 7);
    Vst[dch * VS_STR + tok] =
        (bf16_t)vg[cbase + (size_t)dch * img + h * WIMG + w];
  }
  for (int e = tid; e < S_; e += 256) {                       // polar phi
    int h = e >> 3, w = nw * SPLIT_ + (e & 7);
    phi[e] = polar[(((size_t)b * HIMG + h) * WIMG + w) * 2 + 1];
  }
  for (int e = tid; e < HD_ * 9; e += 256)                    // dw-conv weights
    cw[e] = convw[(size_t)(head * HD_) * 9 + e];
  if (tid < HD_) cb[tid] = convb[head * HD_ + tid];
  __syncthreads();

  // ---------------- Phase 2: score = (Q/8)Kt + rpe, f32 to LDS -------------
  {
    const int mt   = wv >> 1;                    // 4 M-stripes, 2 waves each
    const int mrow = mt * 16 + (lane & 15);      // A: lane -> M row
    const bf16_t* qrow = Qs + mrow * QS_STR;
    const int klo = (lane < 16) ? 0 : 8;         // A K-chunk select (ISA layout)
    Frag a0, a1;
    a0.h[0] = *(const v8bf*)(qrow + klo);
    a0.h[1] = *(const v8bf*)(qrow + klo + 16);
    a1.h[0] = *(const v8bf*)(qrow + 32 + klo);
    a1.h[1] = *(const v8bf*)(qrow + 32 + klo + 16);
    const int kb    = (lane >> 4) * 16;          // unused; B loads use klo
    const int mbase = mt * 16 + (lane >> 4) * 8; // C layout: M = mbase + r

    // Hoist loop-invariant row-phi values into registers (the compiler cannot
    // hoist these LDS loads past the score stores due to aliasing).
    float phim[8];
    #pragma unroll
    for (int r = 0; r < 8; ++r) phim[r] = phi[m0 + mbase + r];

    #pragma unroll 2
    for (int j = 0; j < 16; ++j) {
      const int ntile = (wv & 1) * 16 + j;       // 32 N-tiles total
      const int ncol  = ntile * 16 + (lane & 15);
      const bf16_t* krow = Ks + ncol * KS_STR;
      Frag b0, b1;
      b0.h[0] = *(const v8bf*)(krow + klo);
      b0.h[1] = *(const v8bf*)(krow + klo + 16);
      b1.h[0] = *(const v8bf*)(krow + 32 + klo);
      b1.h[1] = *(const v8bf*)(krow + 32 + klo + 16);
      v8f acc = {};
      acc = wmma_bf16(a0, b0, acc);
      acc = wmma_bf16(a1, b1, acc);
      const float phin = phi[ncol];
      float* scol = score + mbase * SC_STR + ncol;
      #pragma unroll
      for (int r = 0; r < 8; ++r)
        scol[r * SC_STR] = acc[r] + phim[r] - phin;   // + rpe
    }
  }
  __syncthreads();

  // ---------------- Phase 3: per-row L1 over keys (vectorized) -------------
  {
    const int row = tid & 63, chunk = tid >> 6;
    const v4f* srow = (const v4f*)(score + row * SC_STR + chunk * 128);
    float s = 0.f;
    #pragma unroll 4
    for (int j = 0; j < 32; ++j) {
      v4f x = srow[j];
      s += fabsf(x.x) + fabsf(x.y) + fabsf(x.z) + fabsf(x.w);
    }
    l1p[chunk * MROWS + row] = s;
  }
  __syncthreads();
  if (tid < MROWS) {
    float l1 = l1p[tid] + l1p[64 + tid] + l1p[128 + tid] + l1p[192 + tid];
    l1f[tid] = PI_HALF / (l1 + 1e-8f);
  }
  __syncthreads();

  // ---------------- Phase 4: activation -> bf16 attn (reuses Ks LDS) -------
  for (int e = tid; e < (MROWS * S_) / 4; e += 256) {
    int row = e >> 7, c4 = (e & 127) * 4;
    v4f sc = *(const v4f*)(score + row * SC_STR + c4);
    const float f = l1f[row];
    v4bf a;
    a.x = (bf16_t)(1.0f - cosf(sc.x * f));
    a.y = (bf16_t)(1.0f - cosf(sc.y * f));
    a.z = (bf16_t)(1.0f - cosf(sc.z * f));
    a.w = (bf16_t)(1.0f - cosf(sc.w * f));
    *(v4bf*)(attn + row * AT_STR + c4) = a;
  }
  __syncthreads();

  // ---------------- Phase 5: out = attn*V + lepe ---------------------------
  {
    const int klo = (lane < 16) ? 0 : 8;
    const int kb  = (lane >> 4) * 16;
    for (int jj = 0; jj < 2; ++jj) {
      const int t2 = wv * 2 + jj;                  // 16 output tiles / block
      const int mt = t2 >> 2, nt = t2 & 3;
      const int arow = mt * 16 + (lane & 15);
      const bf16_t* ar = attn + arow * AT_STR;
      const int dcol = nt * 16 + (lane & 15);      // head channel
      const bf16_t* vr = Vst + dcol * VS_STR;
      v8f acc = {};
      #pragma unroll 4
      for (int kc = 0; kc < 16; ++kc) {            // K = 512 in steps of 32
        const int ka = kc * 32;
        Frag a, bb;
        a.h[0]  = *(const v8bf*)(ar + ka + klo);
        a.h[1]  = *(const v8bf*)(ar + ka + klo + 16);
        bb.h[0] = *(const v8bf*)(vr + ka + klo);
        bb.h[1] = *(const v8bf*)(vr + ka + klo + 16);
        acc = wmma_bf16(a, bb, acc);
      }
      // Epilogue: add depthwise 3x3 lepe, store to [B, H*W, C]
      const int mbase = mt * 16 + (lane >> 4) * 8;
      const int cg = head * HD_ + dcol;
      const float* vchan = vg + ((size_t)b * DIM_ + cg) * img;
      float wtap[9];
      #pragma unroll
      for (int t = 0; t < 9; ++t) wtap[t] = cw[dcol * 9 + t];
      const float bias = cb[dcol];
      #pragma unroll
      for (int r = 0; r < 8; ++r) {
        const int s = m0 + mbase + r;
        const int h = s >> 3, w = nw * SPLIT_ + (s & 7);
        float lepe = bias;
        #pragma unroll
        for (int dy = 0; dy < 3; ++dy) {
          const int hh = h + dy - 1;
          if ((unsigned)hh < HIMG) {
            #pragma unroll
            for (int dx = 0; dx < 3; ++dx) {
              const int ww = w + dx - 1;
              if ((unsigned)ww < WIMG)
                lepe += vchan[hh * WIMG + ww] * wtap[dy * 3 + dx];
            }
          }
        }
        out[((size_t)b * img + (size_t)h * WIMG + w) * DIM_ + cg] = acc[r] + lepe;
      }
    }
  }
}

extern "C" void kernel_launch(void* const* d_in, const int* in_sizes, int n_in,
                              void* d_out, int out_size, void* d_ws, size_t ws_size,
                              hipStream_t stream) {
  (void)in_sizes; (void)n_in; (void)out_size; (void)d_ws; (void)ws_size;
  const float* temp  = (const float*)d_in[0];   // [3, 8, 512, 64, 64]
  const float* polar = (const float*)d_in[1];   // [8, 64, 64, 2]
  const float* convw = (const float*)d_in[2];   // [512, 1, 3, 3]
  const float* convb = (const float*)d_in[3];   // [512]
  float* outp = (float*)d_out;                  // [8, 4096, 512]

  (void)hipFuncSetAttribute((const void*)hsattn_cdna5,
                            hipFuncAttributeMaxDynamicSharedMemorySize,
                            SMEM_BYTES);
  dim3 grid(B_ * (WIMG / SPLIT_), HEADS_, S_ / MROWS);  // (64, 8, 8)
  dim3 block(256);
  hsattn_cdna5<<<grid, block, SMEM_BYTES, stream>>>(temp, polar, convw, convb, outp);
}
